// TransMatch_43550968381714
// MI455X (gfx1250) — matrix operand order repaired and checked by reading, not compiled, not run on hardware
//
#include <hip/hip_runtime.h>

typedef __attribute__((ext_vector_type(16))) _Float16 v16h;
typedef __attribute__((ext_vector_type(8)))  float    v8f;

#define BS 1024
#define DIM 64
#define NS 16
#define TOPK 10

// ---- f16 weight region element offsets inside d_ws ----
#define OFF_REL_E    0        // rel_emb   64*64   = 4096
#define OFF_WENT0_E  4096     // w_ent0   128*64   = 8192
#define OFF_W0_E     12288    // w0       192*64   = 12288
#define OFF_WENT1_E  24576    // w_ent1            = 8192
#define OFF_W1_E     32768    // w1                = 12288
#define OFF_WQ_E     45056    // wq                = 4096
#define OFF_WK_E     49152
#define OFF_WV_E     53248
#define F16_TOTAL_E  57344
#define F32_REGION_BYTES_OFF 131072  // f16 region rounded up

// float-region offsets (in floats) from ws+F32_REGION_BYTES_OFF
#define OFF_V1MEAN 0              // [1024][2][64] = 131072
#define OFF_V0A    131072         // [1024][64]    = 65536
#define OFF_CTX    196608         // [1024][64]
#define OFF_Q      262144         // [1024][64]
#define OFF_K      327680         // [1024*10][64] = 655360
#define OFF_V      983040         // [1024*10][64]

static __device__ __forceinline__ v8f wmma16(v16h a, v16h b, v8f c) {
    return __builtin_amdgcn_wmma_f32_16x16x32_f16(false, a, false, b, (short)0, c,
                                                  false, false);
}

// Async 16B copy global -> LDS (ASYNCcnt-tracked, CDNA5 ISA).
// VDST VGPR carries the LDS byte offset (low 32 bits of the generic LDS address).
static __device__ __forceinline__ void async_b128(void* lds_ptr, const void* gptr) {
    unsigned loff = (unsigned)(unsigned long long)lds_ptr;
    unsigned long long ga = (unsigned long long)gptr;
    asm volatile("global_load_async_to_lds_b128 %0, %1, off"
                 :: "v"(loff), "v"(ga)
                 : "memory");
}
static __device__ __forceinline__ void wait_async0() {
    asm volatile("s_wait_asynccnt 0x0" ::: "memory");
}

// A fragment: 16x32 f16 tile, rows r0..r0+15, K cols k0..k0+31, per ISA layout:
// lanes 0-15 hold K 0-7 (VGPR0-3) and 16-23 (VGPR4-7); lanes 16-31 hold K 8-15 / 24-31.
static __device__ __forceinline__ v16h load_A(const _Float16* A, int lda, int r0,
                                              int k0, int lane) {
    v16h a;
    int M = lane & 15, h = lane >> 4;
    const _Float16* row = A + (r0 + M) * lda;
#pragma unroll
    for (int v = 0; v < 8; ++v) {
        int k = k0 + ((v >> 2) << 4) + (h << 3) + ((v & 3) << 1);
        a[2 * v]     = row[k];
        a[2 * v + 1] = row[k + 1];
    }
    return a;
}

// B fragment: 32x16 f16 tile from row-major [K][64] weights.
// same K layout as A: lanes 0-15 hold K 0-7 / 16-23, lanes 16-31 hold K 8-15 / 24-31.
static __device__ __forceinline__ v16h load_B(const _Float16* W, int k0, int n0,
                                              int lane) {
    v16h b;
    int N = (lane & 15) + n0, h = lane >> 4;
#pragma unroll
    for (int v = 0; v < 8; ++v) {
        int k = k0 + ((v >> 2) << 4) + (h << 3) + ((v & 3) << 1);
        b[2 * v]     = W[k * 64 + N];
        b[2 * v + 1] = W[(k + 1) * 64 + N];
    }
    return b;
}

// C/D 16x16 f32: lane N = lane&15, VGPR r -> row r + 8*(lane>>4)
static __device__ __forceinline__ void store_D_f16(_Float16* O, int ldo, int r0, int n0,
                                                   int lane, v8f c, const float* bias) {
    int N = lane & 15, h = lane >> 4;
    float bn = bias[n0 + N];
#pragma unroll
    for (int r = 0; r < 8; ++r)
        O[(r0 + r + 8 * h) * ldo + n0 + N] = (_Float16)(c[r] + bn);
}

static __device__ __forceinline__ void store_D_f32(float* O, int ldo, int r0, int n0,
                                                   int lane, v8f c, const float* bias) {
    int N = lane & 15, h = lane >> 4;
    float bn = bias[n0 + N];
#pragma unroll
    for (int r = 0; r < 8; ++r)
        O[(size_t)(r0 + r + 8 * h) * ldo + n0 + N] = c[r] + bn;
}

// ---------------- prep: f32 -> f16 conversion of weights + rel table ----------------
__global__ void cvt_f16(const float* rel, const float* we0, const float* w0,
                        const float* we1, const float* w1, const float* wq,
                        const float* wk, const float* wv, _Float16* dst) {
    int i = blockIdx.x * blockDim.x + threadIdx.x;
    if (i >= F16_TOTAL_E) return;
    const float* src; int off;
    if (i < OFF_WENT0_E)      { src = rel; off = OFF_REL_E; }
    else if (i < OFF_W0_E)    { src = we0; off = OFF_WENT0_E; }
    else if (i < OFF_WENT1_E) { src = w0;  off = OFF_W0_E; }
    else if (i < OFF_W1_E)    { src = we1; off = OFF_WENT1_E; }
    else if (i < OFF_WQ_E)    { src = w1;  off = OFF_W1_E; }
    else if (i < OFF_WK_E)    { src = wq;  off = OFF_WQ_E; }
    else if (i < OFF_WV_E)    { src = wk;  off = OFF_WK_E; }
    else                      { src = wv;  off = OFF_WV_E; }
    dst[i] = (_Float16)src[i - off];
}

// ---------------- stage 1: hop-1 (+hop-0) aggregation with aggregator 0 --------------
// One block (4 waves) per batch row. rel table LDS-resident (async copy); two WMMA GEMMs.
__global__ __launch_bounds__(128) void stage1(
    const int* __restrict__ relations, const int* __restrict__ entity_pairs,
    const int* __restrict__ train_edges, const int* __restrict__ entity2edges,
    const int* __restrict__ edge2entities, const int* __restrict__ edge2relation,
    const float* __restrict__ ent_emb, const _Float16* __restrict__ relf16,
    const _Float16* __restrict__ went0, const _Float16* __restrict__ w0f,
    const float* __restrict__ b_ent0, const float* __restrict__ b0,
    float* __restrict__ v1mean, float* __restrict__ v0a) {
    __shared__ _Float16 s_rel[64 * 64];                    // 8 KB
    __shared__ __align__(16) _Float16 s_Anv[80 * 128];     // 20 KB (aliased as f32 out)
    __shared__ _Float16 s_nv[80 * 64];                     // 10 KB
    __shared__ _Float16 s_Asv[48 * 192];                   // 18 KB
    __shared__ int s_edges0[32];
    __shared__ int s_mask0[32];
    __shared__ int s_nodes[66];
    __shared__ int s_rid[66 * 16];
    __shared__ int s_selfrid[33];

    int t = threadIdx.x, lane = t & 31, wave = t >> 5;
    int b = blockIdx.x;
    int te = train_edges[b];

    __builtin_prefetch(went0, 0, 1);
    __builtin_prefetch(w0f, 0, 1);

    // rel_emb f16 -> LDS: 512 x 16B async copies (no VGPR round trip, ASYNCcnt)
    {
        char* lbase = (char*)s_rel;
        const char* gsrc = (const char*)relf16;
#pragma unroll
        for (int j = 0; j < 4; ++j) {
            int c = t + 128 * j;
            async_b128(lbase + c * 16, gsrc + c * 16);
        }
    }
    if (t < 32) {
        int ent = entity_pairs[2 * b + (t >> 4)];
        int e = entity2edges[(size_t)ent * NS + (t & 15)];
        s_edges0[t] = e;
        s_mask0[t] = (e != te) ? 1 : 0;
        s_selfrid[t] = edge2relation[e];
    } else if (t == 32) {
        s_selfrid[32] = relations[b];
    }
    wait_async0();
    __syncthreads();
    if (t < 66) {
        int node = (t < 64) ? edge2entities[2 * (size_t)s_edges0[t >> 1] + (t & 1)]
                            : entity_pairs[2 * b + (t - 64)];
        s_nodes[t] = node;
    }
    __syncthreads();
    for (int q = t; q < 66 * 16; q += 128) {
        int r = q >> 4, s = q & 15;
        int e = (r < 64) ? entity2edges[(size_t)s_nodes[r] * NS + s]
                         : s_edges0[(r - 64) * 16 + s];
        s_rid[q] = (e != te) ? edge2relation[e] : -1;
    }
    __syncthreads();
    // Build A_nv rows: [ent_emb(node) | masked-mean(rel rows)], 66 rows, K=128
    for (int base = 0; base < 66; base += 64) {
        int r = base + (t >> 1);
        if (r < 66) {
            int half = (t & 1) * 32;
            const float* er = ent_emb + (size_t)s_nodes[r] * DIM + half;
            float acc[32];
#pragma unroll
            for (int d = 0; d < 32; ++d) acc[d] = 0.f;
            int cnt = 0;
            for (int s = 0; s < 16; ++s) {
                int rid = s_rid[r * 16 + s];
                if (rid >= 0) {
                    ++cnt;
                    const _Float16* rv = s_rel + rid * 64 + half;
#pragma unroll
                    for (int d = 0; d < 32; ++d) acc[d] += (float)rv[d];
                }
            }
            float inv = 1.f / (float)(cnt > 0 ? cnt : 1);
            _Float16* Ar = s_Anv + r * 128;
#pragma unroll
            for (int d = 0; d < 32; ++d) {
                Ar[half + d] = (_Float16)er[d];
                Ar[64 + half + d] = (_Float16)(acc[d] * inv);
            }
        }
    }
    for (int q = t; q < (80 - 66) * 128; q += 128) s_Anv[66 * 128 + q] = (_Float16)0.f;
    __syncthreads();
    // GEMM 1: [80,128] x [128,64] -> s_nv (f16, +b_ent0). 20 tiles over 4 waves.
    for (int i = 0; i < 5; ++i) {
        int tile = wave + 4 * i;
        int rt = tile >> 2, ct = tile & 3;
        v8f c = {};
#pragma unroll
        for (int k0 = 0; k0 < 128; k0 += 32) {
            v16h a = load_A(s_Anv, 128, rt * 16, k0, lane);
            v16h bm = load_B(went0, k0, ct * 16, lane);
            c = wmma16(a, bm, c);
        }
        store_D_f16(s_nv, 64, rt * 16, ct * 16, lane, c, b_ent0);
    }
    __syncthreads();
    // Build A_sv rows: [ev_self | nv[.,0] | nv[.,1]], 33 rows, K=192; zero-pad to 48
    for (int q = t; q < 48 * 192; q += 128) {
        int r = q / 192, k = q - r * 192;
        _Float16 val = (_Float16)0.f;
        if (r < 33) {
            if (k < 64) {
                val = s_rel[s_selfrid[r] * 64 + k];
            } else {
                int nb = (r < 32) ? 2 * r : 64;
                val = (k < 128) ? s_nv[nb * 64 + (k - 64)]
                                : s_nv[(nb + 1) * 64 + (k - 128)];
            }
        }
        s_Asv[q] = val;
    }
    __syncthreads();
    // GEMM 2: [48,192] x [192,64] -> s_out f32 (+b0); alias output onto s_Anv space
    float* s_out = (float*)s_Anv;
    for (int i = 0; i < 3; ++i) {
        int tile = wave + 4 * i;
        int rt = tile >> 2, ct = tile & 3;
        v8f c = {};
#pragma unroll
        for (int k0 = 0; k0 < 192; k0 += 32) {
            v16h a = load_A(s_Asv, 192, rt * 16, k0, lane);
            v16h bm = load_B(w0f, k0, ct * 16, lane);
            c = wmma16(a, bm, c);
        }
        store_D_f32(s_out, 64, rt * 16, ct * 16, lane, c, b0);
    }
    __syncthreads();
    // Emit masked mean of v1 over each 16-group (stage-2 neighbor input) and v0a
    {
        int p = t >> 6, d = t & 63;
        float acc = 0.f;
        int cnt = 0;
        for (int s = 0; s < 16; ++s) {
            int m = s_mask0[p * 16 + s];
            if (m) { acc += s_out[(p * 16 + s) * 64 + d]; ++cnt; }
        }
        v1mean[(size_t)b * 128 + p * 64 + d] = acc / (float)(cnt > 0 ? cnt : 1);
        if (t < 64) v0a[(size_t)b * 64 + t] = s_out[32 * 64 + t];
    }
}

// ---------------- stage 2: hop-0 aggregation with aggregator 1 -> ctx ---------------
__global__ __launch_bounds__(128) void stage2(
    const int* __restrict__ entity_pairs, const float* __restrict__ ent_emb,
    const _Float16* __restrict__ went1, const _Float16* __restrict__ w1f,
    const float* __restrict__ b_ent1, const float* __restrict__ b1,
    const float* __restrict__ v1mean, const float* __restrict__ v0a,
    float* __restrict__ ctx) {
    __shared__ _Float16 s_Anv[64 * 128];   // 16 KB
    __shared__ _Float16 s_nv[64 * 64];     // 8 KB
    __shared__ _Float16 s_Asv[32 * 192];   // 12 KB
    int t = threadIdx.x, lane = t & 31, wave = t >> 5;
    int b0 = blockIdx.x * 32;

    for (int q = t; q < 64 * 128; q += 128) {
        int r = q >> 7, k = q & 127;
        int b = b0 + (r >> 1), p = r & 1;
        float v = (k < 64) ? ent_emb[(size_t)entity_pairs[2 * b + p] * DIM + k]
                           : v1mean[(size_t)b * 128 + p * 64 + (k - 64)];
        s_Anv[q] = (_Float16)v;
    }
    __syncthreads();
    for (int i = 0; i < 4; ++i) {   // 16 tiles: [64,128]x[128,64]
        int tile = wave + 4 * i;
        int rt = tile >> 2, ct = tile & 3;
        v8f c = {};
#pragma unroll
        for (int k0 = 0; k0 < 128; k0 += 32) {
            v16h a = load_A(s_Anv, 128, rt * 16, k0, lane);
            v16h bm = load_B(went1, k0, ct * 16, lane);
            c = wmma16(a, bm, c);
        }
        store_D_f16(s_nv, 64, rt * 16, ct * 16, lane, c, b_ent1);
    }
    __syncthreads();
    for (int q = t; q < 32 * 192; q += 128) {
        int r = q / 192, k = q - r * 192;
        _Float16 val;
        if (k < 64)       val = (_Float16)v0a[(size_t)(b0 + r) * 64 + k];
        else if (k < 128) val = s_nv[(2 * r) * 64 + (k - 64)];
        else              val = s_nv[(2 * r + 1) * 64 + (k - 128)];
        s_Asv[q] = val;
    }
    __syncthreads();
    for (int i = 0; i < 2; ++i) {   // 8 tiles: [32,192]x[192,64]
        int tile = wave + 4 * i;
        int rt = tile >> 2, ct = tile & 3;
        v8f c = {};
#pragma unroll
        for (int k0 = 0; k0 < 192; k0 += 32) {
            v16h a = load_A(s_Asv, 192, rt * 16, k0, lane);
            v16h bm = load_B(w1f, k0, ct * 16, lane);
            c = wmma16(a, bm, c);
        }
        int N = lane & 15, h = lane >> 4;
        float bn = b1[ct * 16 + N];
#pragma unroll
        for (int r = 0; r < 8; ++r) {
            int row = rt * 16 + r + 8 * h;
            ctx[(size_t)(b0 + row) * 64 + ct * 16 + N] = c[r] + bn;
        }
    }
}

// ---------------- stage 3a: Q projection (user_emb @ wq + bq) ----------------------
__global__ __launch_bounds__(128) void qproj(const int* __restrict__ user_ids,
                                             const float* __restrict__ user_emb,
                                             const _Float16* __restrict__ wqf,
                                             const float* __restrict__ bq,
                                             float* __restrict__ Q) {
    __shared__ _Float16 s_A[64 * 64];
    int t = threadIdx.x, lane = t & 31, wave = t >> 5;
    int r0 = blockIdx.x * 64;
    for (int q = t; q < 64 * 64; q += 128) {
        int r = q >> 6, k = q & 63;
        s_A[q] = (_Float16)user_emb[(size_t)user_ids[r0 + r] * DIM + k];
    }
    __syncthreads();
    for (int i = 0; i < 4; ++i) {
        int tile = wave + 4 * i;
        int rt = tile >> 2, ct = tile & 3;
        v8f c = {};
#pragma unroll
        for (int k0 = 0; k0 < 64; k0 += 32) {
            v16h a = load_A(s_A, 64, rt * 16, k0, lane);
            v16h bm = load_B(wqf, k0, ct * 16, lane);
            c = wmma16(a, bm, c);
        }
        store_D_f32(Q, 64, r0 + rt * 16, ct * 16, lane, c, bq);
    }
}

// ---------------- stage 3a': K/V projections (shared A fragment) -------------------
__global__ __launch_bounds__(128) void kvproj(const int* __restrict__ top_k_ids,
                                              const float* __restrict__ ent_emb,
                                              const _Float16* __restrict__ wkf,
                                              const _Float16* __restrict__ wvf,
                                              const float* __restrict__ bk,
                                              const float* __restrict__ bv,
                                              float* __restrict__ Km,
                                              float* __restrict__ Vm) {
    __shared__ _Float16 s_A[64 * 64];
    int t = threadIdx.x, lane = t & 31, wave = t >> 5;
    int r0 = blockIdx.x * 64;   // 160 blocks * 64 = 10240 rows
    for (int q = t; q < 64 * 64; q += 128) {
        int r = q >> 6, k = q & 63;
        s_A[q] = (_Float16)ent_emb[(size_t)top_k_ids[r0 + r] * DIM + k];
    }
    __syncthreads();
    for (int i = 0; i < 4; ++i) {
        int tile = wave + 4 * i;
        int rt = tile >> 2, ct = tile & 3;
        v8f cK = {}, cV = {};
#pragma unroll
        for (int k0 = 0; k0 < 64; k0 += 32) {
            v16h a = load_A(s_A, 64, rt * 16, k0, lane);
            v16h bK = load_B(wkf, k0, ct * 16, lane);
            cK = wmma16(a, bK, cK);
            v16h bV = load_B(wvf, k0, ct * 16, lane);
            cV = wmma16(a, bV, cV);
        }
        store_D_f32(Km, 64, r0 + rt * 16, ct * 16, lane, cK, bk);
        store_D_f32(Vm, 64, r0 + rt * 16, ct * 16, lane, cV, bv);
    }
}

// ---------------- stage 3b: attention softmax + final score (wave per row) ----------
__global__ __launch_bounds__(256) void score(const float* __restrict__ Q,
                                             const float* __restrict__ Km,
                                             const float* __restrict__ Vm,
                                             const float* __restrict__ ctx,
                                             const float* __restrict__ w_s,
                                             const float* __restrict__ b_s,
                                             float* __restrict__ out) {
    int lane = threadIdx.x & 31;
    int wave = threadIdx.x >> 5;
    int b = blockIdx.x * 8 + wave;
    int d0 = lane * 2;
    float q0 = Q[(size_t)b * 64 + d0], q1 = Q[(size_t)b * 64 + d0 + 1];
    float c0 = ctx[(size_t)b * 64 + d0], c1 = ctx[(size_t)b * 64 + d0 + 1];
    float logits[TOPK];
#pragma unroll
    for (int k = 0; k < TOPK; ++k) {
        const float* kr = Km + ((size_t)b * TOPK + k) * 64 + d0;
        float p = q0 * kr[0] + q1 * kr[1];
#pragma unroll
        for (int off = 16; off > 0; off >>= 1) p += __shfl_xor(p, off, 32);
        logits[k] = p * 0.125f;  // 1/sqrt(64)
    }
    float m = logits[0];
#pragma unroll
    for (int k = 1; k < TOPK; ++k) m = fmaxf(m, logits[k]);
    float e[TOPK], esum = 0.f;
#pragma unroll
    for (int k = 0; k < TOPK; ++k) { e[k] = __expf(logits[k] - m); esum += e[k]; }
    float inv = 1.f / esum, u0 = 0.f, u1 = 0.f;
#pragma unroll
    for (int k = 0; k < TOPK; ++k) {
        const float* vr = Vm + ((size_t)b * TOPK + k) * 64 + d0;
        float a = e[k] * inv;
        u0 += a * vr[0];
        u1 += a * vr[1];
    }
    float s = u0 * c0 + u1 * c1 + c0 * w_s[d0] + c1 * w_s[d0 + 1];
#pragma unroll
    for (int off = 16; off > 0; off >>= 1) s += __shfl_xor(s, off, 32);
    if (lane == 0) out[b] = s + b_s[0];
}

extern "C" void kernel_launch(void* const* d_in, const int* in_sizes, int n_in,
                              void* d_out, int out_size, void* d_ws, size_t ws_size,
                              hipStream_t stream) {
    const int* relations     = (const int*)d_in[0];
    const int* entity_pairs  = (const int*)d_in[1];
    const int* train_edges   = (const int*)d_in[2];
    const int* user_ids      = (const int*)d_in[3];
    const int* top_k_ids     = (const int*)d_in[4];
    const int* entity2edges  = (const int*)d_in[5];
    const int* edge2entities = (const int*)d_in[6];
    const int* edge2relation = (const int*)d_in[7];
    const float* ent_emb  = (const float*)d_in[8];
    const float* rel_emb  = (const float*)d_in[9];
    const float* user_emb = (const float*)d_in[10];
    const float* w_ent0 = (const float*)d_in[11];
    const float* b_ent0 = (const float*)d_in[12];
    const float* w0     = (const float*)d_in[13];
    const float* b0     = (const float*)d_in[14];
    const float* w_ent1 = (const float*)d_in[15];
    const float* b_ent1 = (const float*)d_in[16];
    const float* w1     = (const float*)d_in[17];
    const float* b1     = (const float*)d_in[18];
    const float* wq = (const float*)d_in[19];
    const float* bq = (const float*)d_in[20];
    const float* wk = (const float*)d_in[21];
    const float* bk = (const float*)d_in[22];
    const float* wv = (const float*)d_in[23];
    const float* bv = (const float*)d_in[24];
    const float* w_s = (const float*)d_in[25];
    const float* b_s = (const float*)d_in[26];

    _Float16* f16b = (_Float16*)d_ws;
    float* f32b = (float*)((char*)d_ws + F32_REGION_BYTES_OFF);
    float* v1mean = f32b + OFF_V1MEAN;
    float* v0a    = f32b + OFF_V0A;
    float* ctx    = f32b + OFF_CTX;
    float* Q      = f32b + OFF_Q;
    float* Km     = f32b + OFF_K;
    float* Vm     = f32b + OFF_V;

    cvt_f16<<<(F16_TOTAL_E + 255) / 256, 256, 0, stream>>>(
        rel_emb, w_ent0, w0, w_ent1, w1, wq, wk, wv, f16b);

    stage1<<<BS, 128, 0, stream>>>(
        relations, entity_pairs, train_edges, entity2edges, edge2entities,
        edge2relation, ent_emb, f16b + OFF_REL_E, f16b + OFF_WENT0_E, f16b + OFF_W0_E,
        b_ent0, b0, v1mean, v0a);

    stage2<<<BS / 32, 128, 0, stream>>>(
        entity_pairs, ent_emb, f16b + OFF_WENT1_E, f16b + OFF_W1_E, b_ent1, b1,
        v1mean, v0a, ctx);

    qproj<<<BS / 64, 128, 0, stream>>>(user_ids, user_emb, f16b + OFF_WQ_E, bq, Q);

    kvproj<<<(BS * TOPK) / 64, 128, 0, stream>>>(
        top_k_ids, ent_emb, f16b + OFF_WK_E, f16b + OFF_WV_E, bk, bv, Km, Vm);

    score<<<BS / 8, 256, 0, stream>>>(Q, Km, Vm, ctx, w_s, b_s, (float*)d_out);
}
